// E_GCL_65644280152411
// MI455X (gfx1250) — matrix operand order repaired and checked by reading, not compiled, not run on hardware
//
#include <hip/hip_runtime.h>
#include <hip/hip_bf16.h>

#define DIM 128

typedef __attribute__((ext_vector_type(16))) _Float16 v16h;
typedef __attribute__((ext_vector_type(8)))  _Float16 v8h;
typedef __attribute__((ext_vector_type(8)))  float    v8f;
typedef __attribute__((ext_vector_type(4)))  float    v4f;

union HV { v16h v; v8h h[2]; };

static __device__ __forceinline__ v8f wmma_f16(v16h a, v16h b, v8f c) {
  return __builtin_amdgcn_wmma_f32_16x16x32_f16(false, a, false, b, (short)0, c,
                                                false, false);
}

// ---------------------------------------------------------------- utilities
__global__ void k_zero(float* __restrict__ p, long long n) {
  long long i = (long long)blockIdx.x * blockDim.x + threadIdx.x;
  long long stride = (long long)gridDim.x * blockDim.x;
  for (; i < n; i += stride) p[i] = 0.0f;
}

__global__ void k_cvt(const float* __restrict__ src, _Float16* __restrict__ dst,
                      long long n) {
  long long i = (long long)blockIdx.x * blockDim.x + threadIdx.x;
  long long stride = (long long)gridDim.x * blockDim.x;
  for (; i < n; i += stride) dst[i] = (_Float16)src[i];
}

// e_w1 is [128][257] row-major; split into f16 [128][256] + fp32 last column.
__global__ void k_cvt_w1(const float* __restrict__ w1, _Float16* __restrict__ dst,
                         float* __restrict__ last) {
  int i = blockIdx.x * blockDim.x + threadIdx.x;   // over 128*257
  if (i >= 128 * 257) return;
  int row = i / 257, col = i % 257;
  float v = w1[i];
  if (col < 256) dst[row * 256 + col] = (_Float16)v;
  else           last[row] = v;
}

__global__ void k_radial(const float* __restrict__ coord, const int* __restrict__ ei,
                         long long E, float* __restrict__ cd, float* __restrict__ radial) {
  long long e = (long long)blockIdx.x * blockDim.x + threadIdx.x;
  if (e >= E) return;
  int r = ei[e], c = ei[E + e];
  float dx = coord[3 * r + 0] - coord[3 * c + 0];
  float dy = coord[3 * r + 1] - coord[3 * c + 1];
  float dz = coord[3 * r + 2] - coord[3 * c + 2];
  cd[3 * e + 0] = dx; cd[3 * e + 1] = dy; cd[3 * e + 2] = dz;
  radial[e] = dx * dx + dy * dy + dz * dz;
}

__global__ void k_bn_final(const float* __restrict__ ssum, const float* __restrict__ ssq,
                           const float* __restrict__ g, const float* __restrict__ be,
                           float count, float* __restrict__ osc, float* __restrict__ osh) {
  int f = threadIdx.x;            // 128 threads
  if (f >= DIM) return;
  float mean = ssum[f] / count;
  float var  = ssq[f] / count - mean * mean;
  float rs   = rsqrtf(var + 1e-5f);
  float sc   = g[f] * rs;
  osc[f] = sc;
  osh[f] = be[f] - mean * sc;
}

// ------------------------------------------------- edge GEMM1: K=257 (h|h|radial)
__global__ __launch_bounds__(256)
void k_gemm_edge1(const _Float16* __restrict__ h16, const int* __restrict__ ei,
                  long long E, const float* __restrict__ radial,
                  const _Float16* __restrict__ W16, const float* __restrict__ w1last,
                  const float* __restrict__ bias, float* __restrict__ Xout,
                  float* __restrict__ ssum, float* __restrict__ ssq) {
  __shared__ float s_sum[DIM], s_sq[DIM];
  int tid = threadIdx.x;
  if (tid < DIM) { s_sum[tid] = 0.0f; s_sq[tid] = 0.0f; }
  __syncthreads();
  int lane = tid & 31, wv = tid >> 5;
  int half = lane >> 4, lm = lane & 15;
  long long tile = (long long)blockIdx.x * 8 + wv;
  bool act = (tile * 16) < E;
  if (act) {
    long long mbase = tile * 16;
    long long e = mbase + lm;                 // A row held by this lane
    int ridx = ei[e], cidx = ei[E + e];
    const _Float16* a0 = h16 + (size_t)ridx * DIM;
    const _Float16* a1 = h16 + (size_t)cidx * DIM;
    v8f acc[8] = {};
#pragma unroll
    for (int kt = 0; kt < 8; ++kt) {
      int kA = kt * 32 + half * 8;            // A chunk start within concat(0..255)
      const _Float16* src = (kA < 128) ? (a0 + kA) : (a1 + (kA - 128));
      HV a;
      a.h[0] = *(const v8h*)(src);
      a.h[1] = *(const v8h*)(src + 16);
      int kB = kt * 32 + half * 8;            // B: two 8-K chunks per lane (table order)
#pragma unroll
      for (int nt = 0; nt < 8; ++nt) {
        const _Float16* wp = W16 + (size_t)(nt * 16 + lm) * 256 + kB;
        HV b;
        b.h[0] = *(const v8h*)wp;
        b.h[1] = *(const v8h*)(wp + 16);
        acc[nt] = wmma_f16(a.v, b.v, acc[nt]);
      }
    }
    float rad[8];
#pragma unroll
    for (int r = 0; r < 8; ++r) rad[r] = radial[mbase + r + half * 8];
#pragma unroll
    for (int nt = 0; nt < 8; ++nt) {
      int n = nt * 16 + lm;
      float bn = bias[n], wl = w1last[n];
      float ps = 0.0f, pq = 0.0f;
#pragma unroll
      for (int r = 0; r < 8; ++r) {
        float v = acc[nt][r] + bn + rad[r] * wl;   // rank-1 radial tail + bias
        Xout[(size_t)(mbase + r + half * 8) * DIM + n] = v;
        ps += v; pq += v * v;
      }
      atomicAdd(&s_sum[n], ps);
      atomicAdd(&s_sq[n], pq);
    }
  }
  __syncthreads();
  if (tid < DIM) { atomicAdd(&ssum[tid], s_sum[tid]); atomicAdd(&ssq[tid], s_sq[tid]); }
}

// ---------------------- generic layer GEMM: out = relu(bn(Xin)) @ W^T + b, K=128
__global__ __launch_bounds__(256)
void k_gemm_layer(const float* __restrict__ Xin, const float* __restrict__ bnsc,
                  const float* __restrict__ bnsh, const _Float16* __restrict__ W16,
                  const float* __restrict__ bias, float* __restrict__ Xout,
                  float* __restrict__ ssum, float* __restrict__ ssq, long long M) {
  __shared__ float s_sum[DIM], s_sq[DIM];
  int tid = threadIdx.x;
  if (tid < DIM) { s_sum[tid] = 0.0f; s_sq[tid] = 0.0f; }
  __syncthreads();
  int lane = tid & 31, wv = tid >> 5;
  int half = lane >> 4, lm = lane & 15;
  long long tile = (long long)blockIdx.x * 8 + wv;
  bool act = (tile * 16) < M;
  if (act) {
    long long mbase = tile * 16;
    const float* xp = Xin + (size_t)(mbase + lm) * DIM;
    v8f acc[8] = {};
#pragma unroll
    for (int kt = 0; kt < 4; ++kt) {
      HV a;
#pragma unroll
      for (int c = 0; c < 2; ++c) {
        int k = kt * 32 + half * 8 + c * 16;
        v4f x0 = *(const v4f*)(xp + k),    x1 = *(const v4f*)(xp + k + 4);
        v4f s0 = *(const v4f*)(bnsc + k),  s1 = *(const v4f*)(bnsc + k + 4);
        v4f t0 = *(const v4f*)(bnsh + k),  t1 = *(const v4f*)(bnsh + k + 4);
#pragma unroll
        for (int j = 0; j < 4; ++j) {
          float u = fmaxf(x0[j] * s0[j] + t0[j], 0.0f);
          float w = fmaxf(x1[j] * s1[j] + t1[j], 0.0f);
          a.h[c][j]     = (_Float16)u;
          a.h[c][j + 4] = (_Float16)w;
        }
      }
      int kB = kt * 32 + half * 8;
#pragma unroll
      for (int nt = 0; nt < 8; ++nt) {
        const _Float16* wp = W16 + (size_t)(nt * 16 + lm) * DIM + kB;
        HV b;
        b.h[0] = *(const v8h*)wp;
        b.h[1] = *(const v8h*)(wp + 16);
        acc[nt] = wmma_f16(a.v, b.v, acc[nt]);
      }
    }
#pragma unroll
    for (int nt = 0; nt < 8; ++nt) {
      int n = nt * 16 + lm;
      float bn = bias[n];
      float ps = 0.0f, pq = 0.0f;
#pragma unroll
      for (int r = 0; r < 8; ++r) {
        float v = acc[nt][r] + bn;
        Xout[(size_t)(mbase + r + half * 8) * DIM + n] = v;
        ps += v; pq += v * v;
      }
      atomicAdd(&s_sum[n], ps);
      atomicAdd(&s_sq[n], pq);
    }
  }
  __syncthreads();
  if (tid < DIM) { atomicAdd(&ssum[tid], s_sum[tid]); atomicAdd(&ssq[tid], s_sq[tid]); }
}

// --------- fused: scale = relu(bn3(Y)) @ c_w2 ; trans scatter ; edge_feat scatter
__global__ __launch_bounds__(256)
void k_scale(const float* __restrict__ Y, const float* __restrict__ sc3,
             const float* __restrict__ sh3, const float* __restrict__ cw2,
             const float* __restrict__ X2, const float* __restrict__ sc2,
             const float* __restrict__ sh2, const float* __restrict__ cd,
             const int* __restrict__ ei, long long E,
             float* __restrict__ trans_sum, float* __restrict__ cnt,
             float* __restrict__ agg_sum) {
  long long gw = ((long long)blockIdx.x * blockDim.x + threadIdx.x) >> 5;
  int lane = threadIdx.x & 31;
  if (gw >= E) return;                     // wave-uniform
  long long e = gw;
  int f = lane * 4;
  v4f y  = *(const v4f*)(Y + (size_t)e * DIM + f);
  v4f s3 = *(const v4f*)(sc3 + f);
  v4f h3 = *(const v4f*)(sh3 + f);
  v4f w2 = *(const v4f*)(cw2 + f);
  float dot = 0.0f;
#pragma unroll
  for (int j = 0; j < 4; ++j) {
    float t = fmaxf(y[j] * s3[j] + h3[j], 0.0f);
    dot += t * w2[j];
  }
#pragma unroll
  for (int off = 16; off > 0; off >>= 1) dot += __shfl_xor(dot, off, 32);
  int row = ei[e];
  // edge_feat = relu(bn2(X2)) scattered into per-node sums
  v4f x2 = *(const v4f*)(X2 + (size_t)e * DIM + f);
  v4f s2 = *(const v4f*)(sc2 + f);
  v4f h2 = *(const v4f*)(sh2 + f);
#pragma unroll
  for (int j = 0; j < 4; ++j) {
    float ef = fmaxf(x2[j] * s2[j] + h2[j], 0.0f);
    atomicAdd(&agg_sum[(size_t)row * DIM + f + j], ef);
  }
  if (lane == 0) {
    float tx = fminf(fmaxf(cd[3 * e + 0] * dot, -100.0f), 100.0f);
    float ty = fminf(fmaxf(cd[3 * e + 1] * dot, -100.0f), 100.0f);
    float tz = fminf(fmaxf(cd[3 * e + 2] * dot, -100.0f), 100.0f);
    atomicAdd(&trans_sum[3 * row + 0], tx);
    atomicAdd(&trans_sum[3 * row + 1], ty);
    atomicAdd(&trans_sum[3 * row + 2], tz);
    atomicAdd(&cnt[row], 1.0f);
  }
}

// ---------------------------- node GEMM1: A = [h | agg_mean], K=256, + BN stats
__global__ __launch_bounds__(256)
void k_node_gemm1(const _Float16* __restrict__ h16, const float* __restrict__ agg,
                  const float* __restrict__ cnt, const _Float16* __restrict__ W16,
                  const float* __restrict__ bias, float* __restrict__ Zout,
                  float* __restrict__ ssum, float* __restrict__ ssq, long long M) {
  __shared__ float s_sum[DIM], s_sq[DIM];
  int tid = threadIdx.x;
  if (tid < DIM) { s_sum[tid] = 0.0f; s_sq[tid] = 0.0f; }
  __syncthreads();
  int lane = tid & 31, wv = tid >> 5;
  int half = lane >> 4, lm = lane & 15;
  long long tile = (long long)blockIdx.x * 8 + wv;
  bool act = (tile * 16) < M;
  if (act) {
    long long mbase = tile * 16;
    long long m = mbase + lm;
    float inv = 1.0f / fmaxf(cnt[m], 1.0f);
    const _Float16* hp = h16 + (size_t)m * DIM;
    const float*    ap = agg + (size_t)m * DIM;
    v8f acc[8] = {};
#pragma unroll
    for (int kt = 0; kt < 8; ++kt) {
      int kA = kt * 32 + half * 8;
      HV a;
      if (kt < 4) {
        a.h[0] = *(const v8h*)(hp + kA);
        a.h[1] = *(const v8h*)(hp + kA + 16);
      } else {
#pragma unroll
        for (int c = 0; c < 2; ++c) {
          int k = kA - 128 + c * 16;
          v4f x0 = *(const v4f*)(ap + k), x1 = *(const v4f*)(ap + k + 4);
#pragma unroll
          for (int j = 0; j < 4; ++j) {
            a.h[c][j]     = (_Float16)(x0[j] * inv);
            a.h[c][j + 4] = (_Float16)(x1[j] * inv);
          }
        }
      }
      int kB = kt * 32 + half * 8;
#pragma unroll
      for (int nt = 0; nt < 8; ++nt) {
        const _Float16* wp = W16 + (size_t)(nt * 16 + lm) * 256 + kB;
        HV b;
        b.h[0] = *(const v8h*)wp;
        b.h[1] = *(const v8h*)(wp + 16);
        acc[nt] = wmma_f16(a.v, b.v, acc[nt]);
      }
    }
#pragma unroll
    for (int nt = 0; nt < 8; ++nt) {
      int n = nt * 16 + lm;
      float bn = bias[n];
      float ps = 0.0f, pq = 0.0f;
#pragma unroll
      for (int r = 0; r < 8; ++r) {
        float v = acc[nt][r] + bn;
        Zout[(size_t)(mbase + r + half * 8) * DIM + n] = v;
        ps += v; pq += v * v;
      }
      atomicAdd(&s_sum[n], ps);
      atomicAdd(&s_sq[n], pq);
    }
  }
  __syncthreads();
  if (tid < DIM) { atomicAdd(&ssum[tid], s_sum[tid]); atomicAdd(&ssq[tid], s_sq[tid]); }
}

// ------------------ node GEMM2: h_out = h + relu(bn(Z1)) @ n_w2^T + b2, K=128
__global__ __launch_bounds__(256)
void k_node_gemm2(const float* __restrict__ Z1, const float* __restrict__ bnsc,
                  const float* __restrict__ bnsh, const _Float16* __restrict__ W16,
                  const float* __restrict__ bias, const float* __restrict__ h,
                  float* __restrict__ out, long long M) {
  int tid = threadIdx.x;
  int lane = tid & 31, wv = tid >> 5;
  int half = lane >> 4, lm = lane & 15;
  long long tile = (long long)blockIdx.x * 8 + wv;
  if ((tile * 16) >= M) return;           // no shared mem: safe early exit
  long long mbase = tile * 16;
  const float* xp = Z1 + (size_t)(mbase + lm) * DIM;
  v8f acc[8] = {};
#pragma unroll
  for (int kt = 0; kt < 4; ++kt) {
    HV a;
#pragma unroll
    for (int c = 0; c < 2; ++c) {
      int k = kt * 32 + half * 8 + c * 16;
      v4f x0 = *(const v4f*)(xp + k),    x1 = *(const v4f*)(xp + k + 4);
      v4f s0 = *(const v4f*)(bnsc + k),  s1 = *(const v4f*)(bnsc + k + 4);
      v4f t0 = *(const v4f*)(bnsh + k),  t1 = *(const v4f*)(bnsh + k + 4);
#pragma unroll
      for (int j = 0; j < 4; ++j) {
        a.h[c][j]     = (_Float16)fmaxf(x0[j] * s0[j] + t0[j], 0.0f);
        a.h[c][j + 4] = (_Float16)fmaxf(x1[j] * s1[j] + t1[j], 0.0f);
      }
    }
    int kB = kt * 32 + half * 8;
#pragma unroll
    for (int nt = 0; nt < 8; ++nt) {
      const _Float16* wp = W16 + (size_t)(nt * 16 + lm) * DIM + kB;
      HV b;
      b.h[0] = *(const v8h*)wp;
      b.h[1] = *(const v8h*)(wp + 16);
      acc[nt] = wmma_f16(a.v, b.v, acc[nt]);
    }
  }
#pragma unroll
  for (int nt = 0; nt < 8; ++nt) {
    int n = nt * 16 + lm;
    float bn = bias[n];
#pragma unroll
    for (int r = 0; r < 8; ++r) {
      size_t idx = (size_t)(mbase + r + half * 8) * DIM + n;
      out[idx] = h[idx] + acc[nt][r] + bn;
    }
  }
}

__global__ void k_coord_out(const float* __restrict__ coord,
                            const float* __restrict__ trans_sum,
                            const float* __restrict__ cnt,
                            float* __restrict__ out, long long Nn) {
  long long i = (long long)blockIdx.x * blockDim.x + threadIdx.x;
  if (i >= Nn) return;
  float inv = 1.0f / fmaxf(cnt[i], 1.0f);
  out[3 * i + 0] = coord[3 * i + 0] + trans_sum[3 * i + 0] * inv;
  out[3 * i + 1] = coord[3 * i + 1] + trans_sum[3 * i + 1] * inv;
  out[3 * i + 2] = coord[3 * i + 2] + trans_sum[3 * i + 2] * inv;
}

// ----------------------------------------------------------------- launcher
extern "C" void kernel_launch(void* const* d_in, const int* in_sizes, int n_in,
                              void* d_out, int out_size, void* d_ws, size_t ws_size,
                              hipStream_t stream) {
  const float* h      = (const float*)d_in[0];
  const float* coord  = (const float*)d_in[1];
  const int*   ei     = (const int*)d_in[2];
  const float* e_w1   = (const float*)d_in[3];
  const float* e_b1   = (const float*)d_in[4];
  const float* e_g1   = (const float*)d_in[5];
  const float* e_be1  = (const float*)d_in[6];
  const float* e_w2   = (const float*)d_in[7];
  const float* e_b2   = (const float*)d_in[8];
  const float* e_g2   = (const float*)d_in[9];
  const float* e_be2  = (const float*)d_in[10];
  const float* n_w1   = (const float*)d_in[11];
  const float* n_b1   = (const float*)d_in[12];
  const float* n_g1   = (const float*)d_in[13];
  const float* n_be1  = (const float*)d_in[14];
  const float* n_w2   = (const float*)d_in[15];
  const float* n_b2   = (const float*)d_in[16];
  const float* c_w1   = (const float*)d_in[17];
  const float* c_b1   = (const float*)d_in[18];
  const float* c_g1   = (const float*)d_in[19];
  const float* c_be1  = (const float*)d_in[20];
  const float* c_w2   = (const float*)d_in[21];

  const long long Nn = in_sizes[0] / DIM;
  const long long E  = in_sizes[2] / 2;

  // ---- workspace layout (256B-aligned bump allocator) ----
  size_t off = 0;
  auto alloc = [&](size_t bytes) -> size_t {
    size_t r = off;
    off = (off + bytes + 255) & ~(size_t)255;
    return r;
  };
  char* ws = (char*)d_ws;
  // zeroed region: stats | agg_sum | trans_sum | cnt
  size_t o_stats = alloc(4 * 2 * DIM * sizeof(float));     // 4 BNs x (sum,sq)
  size_t o_agg   = alloc((size_t)Nn * DIM * sizeof(float));
  size_t o_trans = alloc((size_t)Nn * 3 * sizeof(float));
  size_t o_cnt   = alloc((size_t)Nn * sizeof(float));
  size_t zero_end = off;
  size_t o_bnp   = alloc(4 * 2 * DIM * sizeof(float));     // 4 BNs x (scale,shift)
  size_t o_rad   = alloc((size_t)E * sizeof(float));
  size_t o_cd    = alloc((size_t)E * 3 * sizeof(float));
  size_t o_h16   = alloc((size_t)Nn * DIM * sizeof(_Float16));
  size_t o_w116  = alloc((size_t)DIM * 256 * sizeof(_Float16));
  size_t o_w1l   = alloc((size_t)DIM * sizeof(float));
  size_t o_w216  = alloc((size_t)DIM * DIM * sizeof(_Float16));
  size_t o_cw116 = alloc((size_t)DIM * DIM * sizeof(_Float16));
  size_t o_nw116 = alloc((size_t)DIM * 256 * sizeof(_Float16));
  size_t o_nw216 = alloc((size_t)DIM * DIM * sizeof(_Float16));
  size_t o_X1    = alloc((size_t)E * DIM * sizeof(float)); // also reused as Y
  size_t o_X2    = alloc((size_t)E * DIM * sizeof(float));
  size_t o_Z1    = alloc((size_t)Nn * DIM * sizeof(float));
  (void)ws_size;

  float*     stats = (float*)(ws + o_stats);
  float*     agg   = (float*)(ws + o_agg);
  float*     trans = (float*)(ws + o_trans);
  float*     cnt   = (float*)(ws + o_cnt);
  float*     bnp   = (float*)(ws + o_bnp);
  float*     rad   = (float*)(ws + o_rad);
  float*     cd    = (float*)(ws + o_cd);
  _Float16*  h16   = (_Float16*)(ws + o_h16);
  _Float16*  w116  = (_Float16*)(ws + o_w116);
  float*     w1l   = (float*)(ws + o_w1l);
  _Float16*  w216  = (_Float16*)(ws + o_w216);
  _Float16*  cw116 = (_Float16*)(ws + o_cw116);
  _Float16*  nw116 = (_Float16*)(ws + o_nw116);
  _Float16*  nw216 = (_Float16*)(ws + o_nw216);
  float*     X1    = (float*)(ws + o_X1);
  float*     X2    = (float*)(ws + o_X2);
  float*     Y     = X1;  // X1 dead once X2 is produced
  float*     Z1    = (float*)(ws + o_Z1);

  float* out_h = (float*)d_out;
  float* out_c = out_h + (size_t)Nn * DIM;

  // 1. zero accumulators
  long long nzero = (long long)(zero_end - o_stats) / sizeof(float);
  k_zero<<<dim3((unsigned)((nzero + 255) / 256)), dim3(256), 0, stream>>>(
      (float*)(ws + o_stats), nzero);

  // 2. precision conversion of weights + h
  k_cvt<<<dim3((unsigned)((Nn * DIM + 255) / 256)), dim3(256), 0, stream>>>(h, h16, Nn * DIM);
  k_cvt_w1<<<dim3((128 * 257 + 255) / 256), dim3(256), 0, stream>>>(e_w1, w116, w1l);
  k_cvt<<<dim3(64), dim3(256), 0, stream>>>(e_w2, w216, DIM * DIM);
  k_cvt<<<dim3(64), dim3(256), 0, stream>>>(c_w1, cw116, DIM * DIM);
  k_cvt<<<dim3(128), dim3(256), 0, stream>>>(n_w1, nw116, DIM * 256);
  k_cvt<<<dim3(64), dim3(256), 0, stream>>>(n_w2, nw216, DIM * DIM);

  // 3. coord_diff + radial
  k_radial<<<dim3((unsigned)((E + 255) / 256)), dim3(256), 0, stream>>>(coord, ei, E, cd, rad);

  unsigned eblocks = (unsigned)(((E + 15) / 16 + 7) / 8);
  unsigned nblocks = (unsigned)(((Nn + 15) / 16 + 7) / 8);

  // 4. edge GEMM1 (+BN1 stats)
  k_gemm_edge1<<<dim3(eblocks), dim3(256), 0, stream>>>(
      h16, ei, E, rad, w116, w1l, e_b1, X1, stats + 0, stats + 128);
  k_bn_final<<<dim3(1), dim3(128), 0, stream>>>(stats + 0, stats + 128, e_g1, e_be1,
                                                (float)E, bnp + 0, bnp + 128);
  // 5. edge GEMM2 (+BN2 stats)
  k_gemm_layer<<<dim3(eblocks), dim3(256), 0, stream>>>(
      X1, bnp + 0, bnp + 128, w216, e_b2, X2, stats + 256, stats + 384, E);
  k_bn_final<<<dim3(1), dim3(128), 0, stream>>>(stats + 256, stats + 384, e_g2, e_be2,
                                                (float)E, bnp + 256, bnp + 384);
  // 6. coord GEMM (+BN3 stats); Y reuses X1 storage
  k_gemm_layer<<<dim3(eblocks), dim3(256), 0, stream>>>(
      X2, bnp + 256, bnp + 384, cw116, c_b1, Y, stats + 512, stats + 640, E);
  k_bn_final<<<dim3(1), dim3(128), 0, stream>>>(stats + 512, stats + 640, c_g1, c_be1,
                                                (float)E, bnp + 512, bnp + 640);
  // 7. fused scale/trans/edge_feat scatter (one wave per edge)
  k_scale<<<dim3((unsigned)((E * 32 + 255) / 256)), dim3(256), 0, stream>>>(
      Y, bnp + 512, bnp + 640, c_w2, X2, bnp + 256, bnp + 384, cd, ei, E,
      trans, cnt, agg);
  // 8. node GEMM1 (+node BN stats)
  k_node_gemm1<<<dim3(nblocks), dim3(256), 0, stream>>>(
      h16, agg, cnt, nw116, n_b1, Z1, stats + 768, stats + 896, Nn);
  k_bn_final<<<dim3(1), dim3(128), 0, stream>>>(stats + 768, stats + 896, n_g1, n_be1,
                                                (float)Nn, bnp + 768, bnp + 896);
  // 9. node GEMM2 with residual -> h_out
  k_node_gemm2<<<dim3(nblocks), dim3(256), 0, stream>>>(
      Z1, bnp + 768, bnp + 896, nw216, n_b2, h, out_h, Nn);
  // 10. coord_out
  k_coord_out<<<dim3((unsigned)((Nn + 255) / 256)), dim3(256), 0, stream>>>(
      coord, trans, cnt, out_c, Nn);
}
